// EfficientAttention_25993142075706
// MI455X (gfx1250) — matrix operand order repaired and checked by reading, not compiled, not run on hardware
//
#include <hip/hip_runtime.h>
#include <math.h>

typedef __bf16 bf16;
typedef __attribute__((ext_vector_type(16))) __bf16 v16bf;
typedef __attribute__((ext_vector_type(8)))  float  v8f;

#define NB 4
#define CIN 512
#define CM 64
#define KC 256
#define NTOK 12544      // 16*28*28
#define HEADS 8
#define TOTAL_OUT 25690112  // 4*512*12544

static __device__ __forceinline__ v8f wmma_bf16(v16bf a, v16bf b, v8f c) {
  return __builtin_amdgcn_wmma_f32_16x16x32_bf16(false, a, false, b, (short)0, c, false, false);
}

// A-fragment element->K mapping (contiguous: k = 16*hi + e, matches B fills)
static __device__ __forceinline__ int kaidx(int e, int hi) {
  return e + (hi << 4);
}

// ---------------- prep: weight frag-swizzle to bf16 + zero accumulators ----------------
static __device__ __forceinline__ void frag_prep(bf16* dst, const float* src, int K, int idx) {
  int e    = idx & 15;
  int lane = (idx >> 4) & 31;
  int mtks = idx >> 9;
  int KS   = K >> 5;
  int mt   = mtks / KS;
  int ks   = mtks - mt * KS;
  int m    = mt * 16 + (lane & 15);
  int k    = ks * 32 + kaidx(e, lane >> 4);
  dst[idx] = (bf16)src[(size_t)m * K + k];
}

__global__ void k_prep(const float* __restrict__ w1, const float* __restrict__ wk,
                       const float* __restrict__ wq, const float* __restrict__ wv,
                       bf16* __restrict__ w1f, bf16* __restrict__ wkf,
                       bf16* __restrict__ wqf, bf16* __restrict__ wvf,
                       float* __restrict__ ctxacc, float* __restrict__ normacc) {
  int idx = blockIdx.x * 256 + threadIdx.x;
  if (idx < 32768)                frag_prep(w1f, w1, 512, idx);
  else if (idx < 49152)           frag_prep(wkf, wk, 64, idx - 32768);
  else if (idx < 65536)           frag_prep(wqf, wq, 64, idx - 49152);
  else if (idx < 81920)           frag_prep(wvf, wv, 64, idx - 65536);
  else if (idx < 81920 + 32768)   ctxacc[idx - 81920] = 0.f;
  else if (idx < 81920 + 32770)   normacc[idx - 114688] = 0.f;
}

// ---------------- GEMM1: mid1 = bf16(W1[64x512] @ x[n][512xN] + b1) ----------------
__global__ void __launch_bounds__(256) k_gemm_pw1(const float* __restrict__ x,
                                                  const bf16* __restrict__ w1f,
                                                  const float* __restrict__ b1,
                                                  bf16* __restrict__ mid1) {
  int n = blockIdx.y;
  int wave = threadIdx.x >> 5, lane = threadIdx.x & 31;
  int hi = lane >> 4, lo = lane & 15;
  int col = blockIdx.x * 128 + wave * 16 + lo;
  v8f acc[4] = {};
  for (int ks = 0; ks < 16; ++ks) {
    const float* xp = x + ((size_t)n * CIN + ks * 32 + hi * 16) * NTOK + col;
    v16bf b;
#pragma unroll
    for (int e = 0; e < 16; ++e) b[e] = (bf16)xp[(size_t)e * NTOK];
#pragma unroll
    for (int mt = 0; mt < 4; ++mt) {
      v16bf a = *reinterpret_cast<const v16bf*>(w1f + ((size_t)(mt * 16 + ks) * 32 + lane) * 16);
      acc[mt] = wmma_bf16(a, b, acc[mt]);
    }
  }
#pragma unroll
  for (int mt = 0; mt < 4; ++mt)
#pragma unroll
    for (int r = 0; r < 8; ++r) {
      int m = mt * 16 + r + hi * 8;
      mid1[((size_t)n * CM + m) * NTOK + col] = (bf16)(acc[mt][r] + b1[m]);
    }
}

// ---------------- depthwise 3x3x3 conv ----------------
__global__ void __launch_bounds__(256) k_dwconv(const bf16* __restrict__ mid1,
                                                const float* __restrict__ dww,
                                                const float* __restrict__ dwb,
                                                bf16* __restrict__ midw) {
  int idx = blockIdx.x * 256 + threadIdx.x;  // exact: 4*64*12544 = 12544*256
  int p = idx % NTOK;
  int nc = idx / NTOK;
  int c = nc & 63;
  int xw = p % 28, y = (p / 28) % 28, t = p / 784;
  const bf16* ip = mid1 + (size_t)nc * NTOK;
  const float* wp = dww + c * 27;
  float s = dwb[c];
#pragma unroll
  for (int dz = 0; dz < 3; ++dz) {
    int tz = t + dz - 1;
    if ((unsigned)tz >= 16u) continue;
#pragma unroll
    for (int dy = 0; dy < 3; ++dy) {
      int yy = y + dy - 1;
      if ((unsigned)yy >= 28u) continue;
#pragma unroll
      for (int dx = 0; dx < 3; ++dx) {
        int xx = xw + dx - 1;
        if ((unsigned)xx >= 28u) continue;
        s += (float)ip[tz * 784 + yy * 28 + xx] * wp[dz * 9 + dy * 3 + dx];
      }
    }
  }
  midw[idx] = (bf16)s;
}

// ---------------- GEMM kqv: out = bf16(W[256x64] @ midw[n][64xN] + b) ----------------
__global__ void __launch_bounds__(256) k_gemm_kqv(const bf16* __restrict__ midw,
                                                  const bf16* __restrict__ wf,
                                                  const float* __restrict__ bias,
                                                  bf16* __restrict__ outp) {
  int n = blockIdx.y;
  int wave = threadIdx.x >> 5, lane = threadIdx.x & 31;
  int mh = wave & 1, cs = wave >> 1;
  int hi = lane >> 4, lo = lane & 15;
  int col = blockIdx.x * 64 + cs * 16 + lo;
  v8f acc[8] = {};
#pragma unroll
  for (int ks = 0; ks < 2; ++ks) {
    const bf16* bp = midw + ((size_t)n * CM + ks * 32 + hi * 16) * NTOK + col;
    v16bf b;
#pragma unroll
    for (int e = 0; e < 16; ++e) b[e] = bp[(size_t)e * NTOK];
#pragma unroll
    for (int j = 0; j < 8; ++j) {
      int mt = mh * 8 + j;
      v16bf a = *reinterpret_cast<const v16bf*>(wf + ((size_t)(mt * 2 + ks) * 32 + lane) * 16);
      acc[j] = wmma_bf16(a, b, acc[j]);
    }
  }
#pragma unroll
  for (int j = 0; j < 8; ++j)
#pragma unroll
    for (int r = 0; r < 8; ++r) {
      int m = mh * 128 + j * 16 + r + hi * 8;
      outp[((size_t)n * KC + m) * NTOK + col] = (bf16)(acc[j][r] + bias[m]);
    }
}

// ---------------- key row stats: max and sum(exp(k - max)) over tokens ----------------
__global__ void __launch_bounds__(256) k_keystats(const bf16* __restrict__ keys,
                                                  float* __restrict__ kmax,
                                                  float* __restrict__ ksum) {
  int row = blockIdx.x;  // n*256 + c
  const bf16* kp = keys + (size_t)row * NTOK;
  __shared__ float red[256];
  float mx = -1e30f;
  for (int i = threadIdx.x; i < NTOK; i += 256) mx = fmaxf(mx, (float)kp[i]);
  red[threadIdx.x] = mx;
  __syncthreads();
  for (int s = 128; s > 0; s >>= 1) {
    if (threadIdx.x < s) red[threadIdx.x] = fmaxf(red[threadIdx.x], red[threadIdx.x + s]);
    __syncthreads();
  }
  mx = red[0];
  __syncthreads();
  float sm = 0.f;
  for (int i = threadIdx.x; i < NTOK; i += 256) sm += __expf((float)kp[i] - mx);
  red[threadIdx.x] = sm;
  __syncthreads();
  for (int s = 128; s > 0; s >>= 1) {
    if (threadIdx.x < s) red[threadIdx.x] += red[threadIdx.x + s];
    __syncthreads();
  }
  if (threadIdx.x == 0) { kmax[row] = mx; ksum[row] = red[0]; }
}

// ---------------- query softmax over hk=32 channels; token-major output ----------------
__global__ void __launch_bounds__(256) k_softq(const bf16* __restrict__ queries,
                                               bf16* __restrict__ softQT) {
  int idx = blockIdx.x * 256 + threadIdx.x;  // exact: 4*8*12544 = 1568*256
  int t = idx % NTOK;
  int nh = idx / NTOK;
  int h = nh & 7, n = nh >> 3;
  const bf16* qp = queries + ((size_t)(n * KC + h * 32)) * NTOK + t;
  float q[32];
  float mx = -1e30f;
#pragma unroll
  for (int j = 0; j < 32; ++j) { q[j] = (float)qp[(size_t)j * NTOK]; mx = fmaxf(mx, q[j]); }
  float sm = 0.f;
#pragma unroll
  for (int j = 0; j < 32; ++j) { q[j] = __expf(q[j] - mx); sm += q[j]; }
  float inv = 1.f / sm;
  bf16* op = softQT + ((size_t)n * NTOK + t) * KC + h * 32;
#pragma unroll
  for (int j = 0; j < 32; ++j) op[j] = (bf16)(q[j] * inv);
}

// ---------------- context GEMM: ctxacc[n][h][kc][v] += exp(K) @ V^T (split-K over t) ----------------
__global__ void __launch_bounds__(256) k_context(const bf16* __restrict__ keys,
                                                 const bf16* __restrict__ values,
                                                 const float* __restrict__ kmax,
                                                 float* __restrict__ ctxacc) {
  int n = blockIdx.y >> 3, h = blockIdx.y & 7;
  int wv = threadIdx.x >> 5, lane = threadIdx.x & 31;
  int slice = blockIdx.x * 8 + wv;  // 0..31
  int hi = lane >> 4, lo = lane & 15;
  v8f acc[2][2] = {};
  float mx[2];
  const bf16* krow[2];
  const bf16* vrow[2];
#pragma unroll
  for (int mt = 0; mt < 2; ++mt) {
    int c = h * 32 + mt * 16 + lo;
    mx[mt] = kmax[n * KC + c];
    krow[mt] = keys + ((size_t)(n * KC + c)) * NTOK;
  }
#pragma unroll
  for (int vt = 0; vt < 2; ++vt) {
    int c = h * 32 + vt * 16 + lo;
    vrow[vt] = values + ((size_t)(n * KC + c)) * NTOK;
  }
  for (int it = slice; it < NTOK / 32; it += 32) {
    int tb = it * 32;
    v16bf a[2], b[2];
#pragma unroll
    for (int mt = 0; mt < 2; ++mt)
#pragma unroll
      for (int e = 0; e < 16; ++e)
        a[mt][e] = (bf16)__expf((float)krow[mt][tb + kaidx(e, hi)] - mx[mt]);
#pragma unroll
    for (int vt = 0; vt < 2; ++vt) {
      const bf16* bp = vrow[vt] + tb + hi * 16;
#pragma unroll
      for (int e = 0; e < 16; ++e) b[vt][e] = bp[e];
    }
#pragma unroll
    for (int mt = 0; mt < 2; ++mt)
#pragma unroll
      for (int vt = 0; vt < 2; ++vt) acc[mt][vt] = wmma_bf16(a[mt], b[vt], acc[mt][vt]);
  }
#pragma unroll
  for (int mt = 0; mt < 2; ++mt)
#pragma unroll
    for (int vt = 0; vt < 2; ++vt)
#pragma unroll
      for (int r = 0; r < 8; ++r) {
        int kc = mt * 16 + r + hi * 8;
        int vc = vt * 16 + lo;
        atomicAdd(&ctxacc[(((size_t)(n * 8 + h) * 32) + kc) * 32 + vc], acc[mt][vt][r]);
      }
}

// ---------------- fold: Mfrag = frag-swizzled bf16( rp_w @ ctx^T / ksum ) ----------------
__global__ void __launch_bounds__(256) k_fold(const float* __restrict__ ctxacc,
                                              const float* __restrict__ rpw,
                                              const float* __restrict__ ksum,
                                              bf16* __restrict__ Mfrag) {
  int idx = blockIdx.x * 256 + threadIdx.x;  // exact: 4<<17 = 2048*256
  int e = idx & 15, lane = (idx >> 4) & 31, ks = (idx >> 9) & 7, mt = (idx >> 12) & 31, n = idx >> 17;
  int o = mt * 16 + (lane & 15);
  int k = ks * 32 + kaidx(e, lane >> 4);  // 0..255  (= h*32 + kc)
  int h = k >> 5, kc = k & 31;
  const float* rw = rpw + (size_t)o * KC + h * 32;
  const float* cx = ctxacc + (((size_t)(n * 8 + h)) * 32 + kc) * 32;
  float s = 0.f;
#pragma unroll
  for (int v = 0; v < 32; ++v) s += rw[v] * cx[v];
  Mfrag[idx] = (bf16)(s / ksum[n * KC + k]);
}

// ---------------- final GEMM: out = M @ softQ + rp_b + residual; fused norm stats ----------------
__global__ void __launch_bounds__(256) k_final(const bf16* __restrict__ Mfrag,
                                               const bf16* __restrict__ softQT,
                                               const float* __restrict__ rpb,
                                               const float* __restrict__ x,
                                               float* __restrict__ outp,
                                               float* __restrict__ normacc) {
  int n = blockIdx.y;
  int wave = threadIdx.x >> 5, lane = threadIdx.x & 31;
  int mg = wave & 3, cs = wave >> 2;
  int hi = lane >> 4, lo = lane & 15;
  int col = blockIdx.x * 32 + cs * 16 + lo;
  v8f acc[8] = {};
#pragma unroll
  for (int ks = 0; ks < 8; ++ks) {
    const bf16* bp = softQT + ((size_t)n * NTOK + col) * KC + ks * 32 + hi * 16;
    v16bf b;
#pragma unroll
    for (int e = 0; e < 16; ++e) b[e] = bp[e];
#pragma unroll
    for (int j = 0; j < 8; ++j) {
      int mt = mg * 8 + j;
      v16bf a = *reinterpret_cast<const v16bf*>(Mfrag + (((size_t)(n * 32 + mt) * 8 + ks) * 32 + lane) * 16);
      acc[j] = wmma_bf16(a, b, acc[j]);
    }
  }
  float lsum = 0.f, lsq = 0.f;
#pragma unroll
  for (int j = 0; j < 8; ++j)
#pragma unroll
    for (int r = 0; r < 8; ++r) {
      int m = mg * 128 + j * 16 + r + hi * 8;
      size_t o = ((size_t)n * CIN + m) * NTOK + col;
      float v = acc[j][r] + rpb[m] + x[o];
      outp[o] = v;
      lsum += v;
      lsq += v * v;
    }
  __shared__ float r0[256], r1[256];
  r0[threadIdx.x] = lsum;
  r1[threadIdx.x] = lsq;
  __syncthreads();
  for (int s = 128; s > 0; s >>= 1) {
    if (threadIdx.x < s) { r0[threadIdx.x] += r0[threadIdx.x + s]; r1[threadIdx.x] += r1[threadIdx.x + s]; }
    __syncthreads();
  }
  if (threadIdx.x == 0) {
    atomicAdd(&normacc[0], r0[0]);
    atomicAdd(&normacc[1], r1[0]);
  }
}

// ---------------- global layer-norm over entire tensor ----------------
__global__ void k_norm(float* __restrict__ outp, const float* __restrict__ normacc) {
  const float cnt = (float)TOTAL_OUT;
  float mean = normacc[0] / cnt;
  float var = normacc[1] / cnt - mean * mean;
  float inv = rsqrtf(var + 1e-5f);
  for (size_t i = (size_t)blockIdx.x * 256 + threadIdx.x; i < (size_t)TOTAL_OUT; i += (size_t)gridDim.x * 256)
    outp[i] = (outp[i] - mean) * inv;
}

extern "C" void kernel_launch(void* const* d_in, const int* in_sizes, int n_in,
                              void* d_out, int out_size, void* d_ws, size_t ws_size,
                              hipStream_t stream) {
  (void)in_sizes; (void)n_in; (void)out_size; (void)ws_size;
  const float* x   = (const float*)d_in[0];
  const float* w1  = (const float*)d_in[1];
  const float* b1  = (const float*)d_in[2];
  const float* dww = (const float*)d_in[3];
  const float* dwb = (const float*)d_in[4];
  const float* kw  = (const float*)d_in[5];
  const float* kb  = (const float*)d_in[6];
  const float* qw  = (const float*)d_in[7];
  const float* qb  = (const float*)d_in[8];
  const float* vw  = (const float*)d_in[9];
  const float* vb  = (const float*)d_in[10];
  const float* rpw = (const float*)d_in[11];
  const float* rpb = (const float*)d_in[12];
  float* out = (float*)d_out;

  char* p = (char*)d_ws;
  auto alloc = [&](size_t bytes) -> char* {
    char* r = p;
    p += (bytes + 255) & ~(size_t)255;
    return r;
  };
  bf16*  w1f     = (bf16*)alloc((size_t)32768 * 2);
  bf16*  wkf     = (bf16*)alloc((size_t)16384 * 2);
  bf16*  wqf     = (bf16*)alloc((size_t)16384 * 2);
  bf16*  wvf     = (bf16*)alloc((size_t)16384 * 2);
  float* ctxacc  = (float*)alloc((size_t)32768 * 4);
  float* normacc = (float*)alloc((size_t)2 * 4);
  float* kmaxb   = (float*)alloc((size_t)1024 * 4);
  float* ksumb   = (float*)alloc((size_t)1024 * 4);
  bf16*  mid1    = (bf16*)alloc((size_t)NB * CM * NTOK * 2);
  bf16*  midw    = (bf16*)alloc((size_t)NB * CM * NTOK * 2);
  bf16*  keys    = (bf16*)alloc((size_t)NB * KC * NTOK * 2);
  bf16*  quer    = (bf16*)alloc((size_t)NB * KC * NTOK * 2);
  bf16*  vals    = (bf16*)alloc((size_t)NB * KC * NTOK * 2);
  bf16*  sqT     = (bf16*)alloc((size_t)NB * NTOK * KC * 2);
  bf16*  Mfrag   = (bf16*)alloc((size_t)524288 * 2);

  k_prep<<<449, 256, 0, stream>>>(w1, kw, qw, vw, w1f, wkf, wqf, wvf, ctxacc, normacc);
  k_gemm_pw1<<<dim3(98, 4), 256, 0, stream>>>(x, w1f, b1, mid1);
  k_dwconv<<<12544, 256, 0, stream>>>(mid1, dww, dwb, midw);
  k_gemm_kqv<<<dim3(196, 4), 256, 0, stream>>>(midw, wkf, kb, keys);
  k_gemm_kqv<<<dim3(196, 4), 256, 0, stream>>>(midw, wqf, qb, quer);
  k_gemm_kqv<<<dim3(196, 4), 256, 0, stream>>>(midw, wvf, vb, vals);
  k_keystats<<<1024, 256, 0, stream>>>(keys, kmaxb, ksumb);
  k_softq<<<1568, 256, 0, stream>>>(quer, sqT);
  k_context<<<dim3(4, 32), 256, 0, stream>>>(keys, vals, kmaxb, ctxacc);
  k_fold<<<2048, 256, 0, stream>>>(ctxacc, rpw, ksumb, Mfrag);
  k_final<<<dim3(392, 4), 256, 0, stream>>>(Mfrag, sqT, rpb, x, out, normacc);
  k_norm<<<4096, 256, 0, stream>>>(out, normacc);
}
